// MultiAgentGRU_41644002902058
// MI455X (gfx1250) — matrix operand order repaired and checked by reading, not compiled, not run on hardware
//
#include <hip/hip_runtime.h>
#include <hip/hip_bf16.h>

typedef __attribute__((ext_vector_type(16))) __bf16 v16bf;
typedef __attribute__((ext_vector_type(8)))  float  v8f;

static constexpr int B_ = 64, S_ = 256, A_ = 8, I_ = 256, H_ = 512;
static constexpr int G3   = 3 * H_;               // 1536
static constexpr long ROWS = (long)B_ * S_ * A_;  // 131072

__device__ __forceinline__ __bf16 f2bf(float f) {
    unsigned u = __builtin_bit_cast(unsigned, f);
    u += 0x7FFFu + ((u >> 16) & 1u);              // round-to-nearest-even
    unsigned short h = (unsigned short)(u >> 16);
    return __builtin_bit_cast(__bf16, h);
}

union FragU { v16bf v; uint4 q[2]; };

// Async copy of one 32B lane-chunk (two b128s, INST_OFFSET applies to both
// LDS and global sides per ISA pseudocode). Tracked on ASYNCcnt.
__device__ __forceinline__ void async_frag_load(unsigned lds_byte, const __bf16* gsrc) {
    unsigned long long ga = (unsigned long long)(uintptr_t)gsrc;
    asm volatile("global_load_async_to_lds_b128 %0, %1, off"
                 :: "v"(lds_byte), "v"(ga) : "memory");
    asm volatile("global_load_async_to_lds_b128 %0, %1, off offset:16"
                 :: "v"(lds_byte), "v"(ga) : "memory");
}

// ---------------------------------------------------------------------------
// Pre-swizzle weight matrix w (N x K, row-major f32) into bf16 WMMA B-operand
// fragments: dst[((ntile*KT + kt)*32 + lane)*16 + j] = w[n*K + k]
//   n = ntile*16 + (lane&15),  k = kt*32 + (lane>=16)*16 + j
// ---------------------------------------------------------------------------
__global__ void swizzle_weights(const float* __restrict__ w, __bf16* __restrict__ dst,
                                int K, int KT, int total) {
    int idx = blockIdx.x * blockDim.x + threadIdx.x;
    if (idx >= total) return;
    int j    = idx & 15;
    int lane = (idx >> 4) & 31;
    int t    = idx >> 9;
    int kt    = t % KT;
    int ntile = t / KT;
    int n = ntile * 16 + (lane & 15);
    int k = kt * 32 + ((lane >> 4) << 4) + j;
    dst[idx] = f2bf(w[(long)n * K + k]);
}

// ---------------------------------------------------------------------------
// Phase 1: gi = x(131072x256) @ w_ih^T(256x1536) + b_ih   (bf16 WMMA, f32 acc)
// One wave per 16-row M-tile; all 8 A-fragments cached in regs (x read once);
// loops 12 N-chunks of 128 cols. gi stored non-temporally (streamed once).
// ---------------------------------------------------------------------------
__global__ void __launch_bounds__(256)
gi_gemm(const float* __restrict__ x, const __bf16* __restrict__ wswz,
        const float* __restrict__ b_ih, float* __restrict__ gi) {
    const int wid  = threadIdx.x >> 5;
    const int lane = threadIdx.x & 31;
    const int m16  = lane & 15;
    const int hi   = lane >> 4;
    const long mtile = (long)blockIdx.x * 8 + wid;
    const float* xr = x + (mtile * 16 + m16) * I_;

    v16bf afr[8];
    #pragma unroll
    for (int kt = 0; kt < 8; ++kt) {
        const float4* p0 = (const float4*)(xr + kt * 32 + hi * 16);
        const float4* p1 = (const float4*)(xr + kt * 32 + hi * 16 + 8);
        float4 f0 = p0[0], f1 = p0[1], f2 = p1[0], f3 = p1[1];
        v16bf a;
        a[0]=f2bf(f0.x); a[1]=f2bf(f0.y); a[2]=f2bf(f0.z); a[3]=f2bf(f0.w);
        a[4]=f2bf(f1.x); a[5]=f2bf(f1.y); a[6]=f2bf(f1.z); a[7]=f2bf(f1.w);
        a[8]=f2bf(f2.x); a[9]=f2bf(f2.y); a[10]=f2bf(f2.z); a[11]=f2bf(f2.w);
        a[12]=f2bf(f3.x); a[13]=f2bf(f3.y); a[14]=f2bf(f3.z); a[15]=f2bf(f3.w);
        afr[kt] = a;
    }

    for (int nc = 0; nc < 12; ++nc) {
        v8f acc[8] = {};
        #pragma unroll
        for (int kt = 0; kt < 8; ++kt) {
            #pragma unroll
            for (int nt = 0; nt < 8; ++nt) {
                const int ntile = nc * 8 + nt;
                FragU bu;
                const uint4* bp = (const uint4*)(wswz + ((ntile * 8 + kt) * 32 + lane) * 16);
                bu.q[0] = bp[0]; bu.q[1] = bp[1];
                acc[nt] = __builtin_amdgcn_wmma_f32_16x16x32_bf16(
                    false, afr[kt], false, bu.v, (short)0, acc[nt], false, false);
            }
        }
        const long orow = mtile * 16;
        #pragma unroll
        for (int nt = 0; nt < 8; ++nt) {
            const int n = (nc * 8 + nt) * 16 + m16;
            const float bias = b_ih[n];
            #pragma unroll
            for (int v = 0; v < 8; ++v) {
                const int m = v + hi * 8;
                __builtin_nontemporal_store(acc[nt][v] + bias,
                                            gi + (orow + m) * (long)G3 + n);
            }
        }
    }
}

// ---------------------------------------------------------------------------
// Phase 2: GRU scan. 32 workgroups x 16 (b,a) rows, 8 waves each.
// w_hh B-fragments double-buffered through LDS with async global->LDS copies.
// ---------------------------------------------------------------------------
static constexpr unsigned HS_OFF  = 0;            // f32 h  : 16*512*4   = 32768
static constexpr unsigned HBF_OFF = 32768;        // bf16 A : 16*32*16*2 = 16384
static constexpr unsigned BST_OFF = 49152;        // B stage: 8*2*12*1024= 196608
static constexpr unsigned SMEM_BYTES = 245760;    // 240 KB  (<= 320 KB/WGP)

__global__ void __launch_bounds__(256)
gru_scan(const float* __restrict__ gi, const __bf16* __restrict__ whh_swz,
         const float* __restrict__ b_hh, const unsigned char* __restrict__ is_init,
         float* __restrict__ out, float* __restrict__ h_n) {
    __shared__ __align__(16) char smem[SMEM_BYTES];
    float*  hs  = (float*)(smem + HS_OFF);        // [16][512]
    __bf16* hbf = (__bf16*)(smem + HBF_OFF);      // [16][32][16]

    const int wid  = threadIdx.x >> 5;
    const int lane = threadIdx.x & 31;
    const int l15  = lane & 15;
    const int hi   = lane >> 4;
    const int rowbase = blockIdx.x * 16;          // global (b*8+a) base
    const int b0 = rowbase >> 3;
    const int wcol = wid * 64;                    // this wave's H-column base

    float bh[4][3];
    #pragma unroll
    for (int ct = 0; ct < 4; ++ct) {
        const int c = wcol + ct * 16 + l15;
        bh[ct][0] = b_hh[c];
        bh[ct][1] = b_hh[512 + c];
        bh[ct][2] = b_hh[1024 + c];
    }

    // Issue all 24 async B-fragment copies for (kt) into staging buffer (buf).
    auto prefetch_kt = [&](int kt, int buf) {
        const unsigned base = BST_OFF + (unsigned)wid * 24576u
                            + (unsigned)buf * 12288u + (unsigned)lane * 32u;
        #pragma unroll
        for (int g = 0; g < 3; ++g) {
            #pragma unroll
            for (int ct = 0; ct < 4; ++ct) {
                const int f = g * 4 + ct;
                const int ntile = g * 32 + wid * 4 + ct;
                async_frag_load(base + (unsigned)f * 1024u,
                                whh_swz + ((ntile * 16 + kt) * 32 + lane) * 16);
            }
        }
    };

    for (int idx = threadIdx.x; idx < 16 * H_; idx += 256)
        hs[idx] = 0.f;

    prefetch_kt(0, 0);                            // prime the pipeline

    for (int t = 0; t < S_; ++t) {
        __syncthreads();                          // prior step's hs writes done
        const unsigned char f0 = is_init[b0 * S_ + t];
        const unsigned char f1 = is_init[(b0 + 1) * S_ + t];
        if (f0 | f1) {
            for (int idx = threadIdx.x; idx < 16 * H_; idx += 256) {
                const int m = idx >> 9;
                if (m < 8 ? f0 : f1) hs[idx] = 0.f;
            }
        }

        // Prefetch this step's gi rows (96 KB) so gate-phase loads hit cache.
        {
            const int l0 = threadIdx.x * 3;
            #pragma unroll
            for (int i = 0; i < 3; ++i) {
                const int l = l0 + i;
                const int m = l / 48;
                const int c = (l % 48) * 32;
                const int row = rowbase + m;
                const long grow = ((long)(row >> 3) * S_ + t) * A_ + (row & 7);
                __builtin_prefetch(gi + grow * G3 + c, 0, 3);
            }
        }
        __syncthreads();

        // Rebuild bf16 A-fragments from f32 state: wave handles kt = 2*wid, +1
        #pragma unroll
        for (int q = 0; q < 2; ++q) {
            const int kt = wid * 2 + q;
            FragU u;
            #pragma unroll
            for (int j = 0; j < 16; ++j) {
                const int k = kt * 32 + hi * 16 + j;
                u.v[j] = f2bf(hs[l15 * H_ + k]);
            }
            *(uint4*)(hbf + (kt * 32 + lane) * 16)     = u.q[0];
            *(uint4*)(hbf + (kt * 32 + lane) * 16 + 8) = u.q[1];
        }
        __syncthreads();

        // gh GEMM: 12 accum tiles = 3 gates x 4 col-tiles owned by this wave.
        // B fragments ping-pong through LDS, one full kt prefetched ahead.
        v8f acc[12] = {};
        for (int kt = 0; kt < 16; ++kt) {
            prefetch_kt((kt + 1) & 15, (kt & 1) ^ 1);   // wraps into next step
            asm volatile("s_wait_asynccnt 0x18" ::: "memory"); // kt's 24 landed
            FragU au;
            au.q[0] = *(const uint4*)(hbf + (kt * 32 + lane) * 16);
            au.q[1] = *(const uint4*)(hbf + (kt * 32 + lane) * 16 + 8);
            const char* bb = smem + BST_OFF + wid * 24576 + (kt & 1) * 12288
                           + lane * 32;
            #pragma unroll
            for (int f = 0; f < 12; ++f) {
                FragU bu;
                const uint4* bp = (const uint4*)(bb + f * 1024);
                bu.q[0] = bp[0]; bu.q[1] = bp[1];
                acc[f] = __builtin_amdgcn_wmma_f32_16x16x32_bf16(
                    false, au.v, false, bu.v, (short)0, acc[f], false, false);
            }
        }

        // Gate math + state update (wave-local columns -> no cross-wave hazard)
        #pragma unroll
        for (int ct = 0; ct < 4; ++ct) {
            const int n = wcol + ct * 16 + l15;
            #pragma unroll
            for (int v = 0; v < 8; ++v) {
                const int m   = v + hi * 8;
                const int row = rowbase + m;
                const long grow = ((long)(row >> 3) * S_ + t) * A_ + (row & 7);
                const float* gir = gi + grow * G3;
                const float gi_r = __builtin_nontemporal_load(gir + n);
                const float gi_z = __builtin_nontemporal_load(gir + 512 + n);
                const float gi_n = __builtin_nontemporal_load(gir + 1024 + n);
                const float ghr = acc[0 * 4 + ct][v] + bh[ct][0];
                const float ghz = acc[1 * 4 + ct][v] + bh[ct][1];
                const float ghn = acc[2 * 4 + ct][v] + bh[ct][2];
                const float r  = 1.f / (1.f + __expf(-(gi_r + ghr)));
                const float z  = 1.f / (1.f + __expf(-(gi_z + ghz)));
                const float nn = tanhf(gi_n + r * ghn);
                const float hold = hs[m * H_ + n];
                const float hnew = (1.f - z) * nn + z * hold;
                hs[m * H_ + n] = hnew;
                __builtin_nontemporal_store(hnew, out + grow * H_ + n);
                if (t == S_ - 1) h_n[(long)row * H_ + n] = hnew;
            }
        }
    }
}

// ---------------------------------------------------------------------------
extern "C" void kernel_launch(void* const* d_in, const int* in_sizes, int n_in,
                              void* d_out, int out_size, void* d_ws, size_t ws_size,
                              hipStream_t stream) {
    const float*         x       = (const float*)d_in[0];
    const unsigned char* is_init = (const unsigned char*)d_in[1];
    const float*         w_ih    = (const float*)d_in[2];
    const float*         w_hh    = (const float*)d_in[3];
    const float*         b_ih    = (const float*)d_in[4];
    const float*         b_hh    = (const float*)d_in[5];
    float* out = (float*)d_out;

    char* ws = (char*)d_ws;
    __bf16* wih_swz = (__bf16*)ws;                         // 1536*256*2  =   786432 B
    __bf16* whh_swz = (__bf16*)(ws + 786432);              // 1536*512*2  =  1572864 B
    float*  gi      = (float*)(ws + 786432 + 1572864);     // 131072*1536*4 bytes

    swizzle_weights<<<(393216 + 255) / 256, 256, 0, stream>>>(w_ih, wih_swz, I_, 8,  393216);
    swizzle_weights<<<(786432 + 255) / 256, 256, 0, stream>>>(w_hh, whh_swz, H_, 16, 786432);
    gi_gemm<<<1024, 256, 0, stream>>>(x, wih_swz, b_ih, gi);
    gru_scan<<<32, 256, 0, stream>>>(gi, whh_swz, b_hh, is_init,
                                     out, out + (size_t)ROWS * H_);
}
